// DimeNetPlusPlusModel_45320494907959
// MI455X (gfx1250) — matrix operand order repaired and checked by reading, not compiled, not run on hardware
//
#include <hip/hip_runtime.h>
#include <hip/hip_bf16.h>
#include <math.h>

// ---------------------------------------------------------------------------
// DimeNet++ forward for MI455X (gfx1250).
// Matrix layers use v_wmma_f32_16x16x32_f16 (f16 inputs, f32 accumulate).
// LDS panels are swizzled so every WMMA fragment loads as 2x ds_load_b128.
// Input order assumption: pos, params pytree leaves in sorted-key order,
// then z, batch, i, j, idx_i, idx_j, idx_k, idx_kj, idx_ji.
// ---------------------------------------------------------------------------

typedef __attribute__((ext_vector_type(16))) _Float16 v16h;
typedef __attribute__((ext_vector_type(8)))  _Float16 v8h;
typedef __attribute__((ext_vector_type(4)))  _Float16 v4h;
typedef __attribute__((ext_vector_type(8)))  float    v8f;

#define HD        128
#define NRR       6
#define NSS       7
#define SBF_DIM   42      // NSS*NRR
#define BAS       8
#define IEMB      64
#define OEMB      256
#define NBLK      4
#define NTGT      12
#define NGRAPH    128
#define INV_CUT   0.1f    // 1/CUTOFF
#define INV4PI    0.07957747154594767f

// Zeros of spherical Bessel functions j_l, l=0..6, first 6 each.
__device__ __constant__ float c_Z[NSS][NRR] = {
  { 3.14159265f,  6.28318531f,  9.42477796f, 12.56637061f, 15.70796327f, 18.84955592f},
  { 4.49340946f,  7.72525184f, 10.90412166f, 14.06619391f, 17.22075527f, 20.37130296f},
  { 5.76345920f,  9.09501133f, 12.32294097f, 15.51460301f, 18.68903635f, 21.85387422f},
  { 6.98793200f, 10.41711855f, 13.69802315f, 16.92362129f, 20.12180620f, 23.30424700f},
  { 8.18256145f, 11.70490710f, 15.03966480f, 18.30125590f, 21.52541770f, 24.72756570f},
  { 9.35581211f, 12.96653010f, 16.35470970f, 19.65315210f, 22.90455060f, 26.12775010f},
  {10.51283540f, 14.20739250f, 17.64797490f, 20.98346310f, 24.26276800f, 27.50786840f}
};

__device__ __forceinline__ float silu_f(float v){ return v / (1.0f + __expf(-v)); }

// --------------------------- utility kernels -------------------------------

__global__ void zero_kernel(float* __restrict__ p, long n){
  long i = blockIdx.x * (long)blockDim.x + threadIdx.x;
  long stride = (long)gridDim.x * blockDim.x;
  for (; i < n; i += stride) p[i] = 0.0f;
}

// norm[l][r] = 1/sqrt(0.5 * j_{l+1}(Z[l][r])^2)
__global__ void init_norm_kernel(float* __restrict__ norm){
  int t = threadIdx.x;
  if (t >= NSS * NRR) return;
  int l = t / NRR;
  float z  = c_Z[l][t % NRR];
  float j0 = sinf(z) / z;
  float jc = sinf(z) / (z * z) - cosf(z) / z;   // j1
  float jm = j0;
  for (int ll = 1; ll <= l; ++ll) {             // after loop: jc = j_{l+1}
    float jn = (2.0f * ll + 1.0f) / z * jc - jm;
    jm = jc; jc = jn;
  }
  norm[t] = 1.0f / sqrtf(0.5f * jc * jc);
}

// Per-edge distance, envelope, Bessel RBF (E x 6) and spherical radial (E x 42).
__global__ void edge_rbf_kernel(const float* __restrict__ pos,
                                const int* __restrict__ ei, const int* __restrict__ ej,
                                const float* __restrict__ freq, const float* __restrict__ norm,
                                float* __restrict__ rbf, float* __restrict__ rbf_s, int E){
  int e = blockIdx.x * blockDim.x + threadIdx.x;
  if (e >= E) return;
  int a = ei[e], b = ej[e];
  float dx = pos[a*3+0] - pos[b*3+0];
  float dy = pos[a*3+1] - pos[b*3+1];
  float dz = pos[a*3+2] - pos[b*3+2];
  float dist = sqrtf(dx*dx + dy*dy + dz*dz);
  float x = dist * INV_CUT;
  float env = 0.0f;
  if (x < 1.0f) {
    float x2 = x * x;
    float x5 = x2 * x2 * x;                     // x^(p-1), p = 6
    env = 1.0f / x - 28.0f * x5 + 48.0f * x5 * x - 21.0f * x5 * x2;
  }
  #pragma unroll
  for (int r = 0; r < NRR; ++r)
    rbf[(long)e*NRR + r] = env * sinf(freq[r] * x);

  for (int l = 0; l < NSS; ++l) {
    for (int r = 0; r < NRR; ++r) {
      float arg = c_Z[l][r] * x;
      float j0 = sinf(arg) / arg;
      float jv;
      if (l == 0) jv = j0;
      else {
        float jm = j0;
        float jc = sinf(arg) / (arg * arg) - cosf(arg) / arg;
        for (int ll = 1; ll < l; ++ll) {
          float jn = (2.0f * ll + 1.0f) / arg * jc - jm;
          jm = jc; jc = jn;
        }
        jv = jc;
      }
      rbf_s[(long)e*SBF_DIM + l*NRR + r] = norm[l*NRR + r] * jv * env;
    }
  }
}

// cat = [ hz[z[i]] | hz[z[j]] | silu(rbf @ emb_rbf_w + b) ]   (E x 384)
__global__ void build_cat_kernel(const float* __restrict__ rbf,
                                 const int* __restrict__ zarr,
                                 const int* __restrict__ ei, const int* __restrict__ ej,
                                 const float* __restrict__ emb_z,
                                 const float* __restrict__ erw, const float* __restrict__ erb,
                                 float* __restrict__ cat, int E){
  long idx = blockIdx.x * (long)blockDim.x + threadIdx.x;
  if (idx >= (long)E * 384) return;
  int e = (int)(idx / 384);
  int c = (int)(idx % 384);
  float v;
  if (c < HD)            v = emb_z[(long)zarr[ei[e]] * HD + c];
  else if (c < 2*HD)     v = emb_z[(long)zarr[ej[e]] * HD + (c - HD)];
  else {
    int cc = c - 2*HD;
    float acc = erb[cc];
    #pragma unroll
    for (int r = 0; r < NRR; ++r) acc += rbf[(long)e*NRR + r] * erw[r*HD + cc];
    v = silu_f(acc);
  }
  cat[idx] = v;
}

// x_kj[e][c] *= (rbf[e] @ rbf1 @ rbf2)[c]
__global__ void gate_xkj_kernel(const float* __restrict__ rbf,
                                const float* __restrict__ rbf1, const float* __restrict__ rbf2,
                                float* __restrict__ xkj, int E){
  long idx = blockIdx.x * (long)blockDim.x + threadIdx.x;
  if (idx >= (long)E * HD) return;
  int e = (int)(idx / HD);
  int c = (int)(idx % HD);
  float r8[BAS];
  #pragma unroll
  for (int b = 0; b < BAS; ++b) {
    float a = 0.0f;
    #pragma unroll
    for (int r = 0; r < NRR; ++r) a += rbf[(long)e*NRR + r] * rbf1[r*BAS + b];
    r8[b] = a;
  }
  float g = 0.0f;
  #pragma unroll
  for (int b = 0; b < BAS; ++b) g += r8[b] * rbf2[b*HD + c];
  xkj[idx] *= g;
}

// Output-block edge term: tn[i[e]][c] += (rbf[e] @ out_rbf_w)[c] * x[e][c]
__global__ void out_scatter_kernel(const float* __restrict__ rbf,
                                   const float* __restrict__ x,
                                   const float* __restrict__ orw,
                                   const int* __restrict__ ei,
                                   float* __restrict__ tn, int E){
  long idx = blockIdx.x * (long)blockDim.x + threadIdx.x;
  if (idx >= (long)E * HD) return;
  int e = (int)(idx / HD);
  int c = (int)(idx % HD);
  float acc = 0.0f;
  #pragma unroll
  for (int r = 0; r < NRR; ++r) acc += rbf[(long)e*NRR + r] * orw[r*HD + c];
  atomicAdd(&tn[(long)ei[e]*HD + c], acc * x[idx]);
}

// Per-triplet: angle -> Legendre cbf, sbf = rbf_s[kj]*cbf, project through
// sbf1 (42x8) and sbf2 (8x64), multiply with x_down[kj], scatter-add to edge ji.
__global__ void triplet_kernel(const float* __restrict__ pos,
                               const int* __restrict__ ii, const int* __restrict__ ij,
                               const int* __restrict__ ik,
                               const int* __restrict__ ikj, const int* __restrict__ iji,
                               const float* __restrict__ rbf_s,
                               const float* __restrict__ sbf1, const float* __restrict__ sbf2,
                               const float* __restrict__ xdown,
                               float* __restrict__ tacc, int T){
  __shared__ float s1[SBF_DIM * BAS];   // 336
  __shared__ float s2[BAS * IEMB];      // 512
  for (int t = threadIdx.x; t < SBF_DIM * BAS; t += blockDim.x) s1[t] = sbf1[t];
  for (int t = threadIdx.x; t < BAS * IEMB;   t += blockDim.x) s2[t] = sbf2[t];
  __syncthreads();

  int t = blockIdx.x * blockDim.x + threadIdx.x;
  if (t >= T) return;
  int ai = ii[t], aj = ij[t], ak = ik[t];
  float jx = pos[aj*3+0] - pos[ai*3+0];
  float jy = pos[aj*3+1] - pos[ai*3+1];
  float jz = pos[aj*3+2] - pos[ai*3+2];
  float kx = pos[ak*3+0] - pos[aj*3+0];
  float ky = pos[ak*3+1] - pos[aj*3+1];
  float kz = pos[ak*3+2] - pos[aj*3+2];
  float dv = jx*kx + jy*ky + jz*kz;
  float cx = jy*kz - jz*ky;
  float cy = jz*kx - jx*kz;
  float cz = jx*ky - jy*kx;
  float bb = sqrtf(cx*cx + cy*cy + cz*cz);
  float cth = cosf(atan2f(bb, dv));

  float P[NSS];
  P[0] = 1.0f; P[1] = cth;
  #pragma unroll
  for (int l = 1; l < NSS - 1; ++l)
    P[l+1] = ((2.0f*l + 1.0f) * cth * P[l] - (float)l * P[l-1]) / (float)(l + 1);
  float cbf[NSS];
  #pragma unroll
  for (int l = 0; l < NSS; ++l) cbf[l] = P[l] * sqrtf((2.0f*l + 1.0f) * INV4PI);

  int ekj = ikj[t], eji = iji[t];
  const float* rs = rbf_s + (long)ekj * SBF_DIM;
  float s8[BAS];
  #pragma unroll
  for (int b = 0; b < BAS; ++b) s8[b] = 0.0f;
  #pragma unroll
  for (int l = 0; l < NSS; ++l) {
    float cb = cbf[l];
    #pragma unroll
    for (int r = 0; r < NRR; ++r) {
      float sv = rs[l*NRR + r] * cb;
      #pragma unroll
      for (int b = 0; b < BAS; ++b) s8[b] += sv * s1[(l*NRR + r)*BAS + b];
    }
  }
  const float* xd = xdown + (long)ekj * IEMB;
  float* op = tacc + (long)eji * IEMB;
  for (int c = 0; c < IEMB; ++c) {
    float v = 0.0f;
    #pragma unroll
    for (int b = 0; b < BAS; ++b) v += s8[b] * s2[b*IEMB + c];
    atomicAdd(&op[c], v * xd[c]);
  }
}

__global__ void graph_pool_kernel(const float* __restrict__ P,
                                  const int* __restrict__ batch,
                                  float* __restrict__ Pg, int NN){
  long idx = blockIdx.x * (long)blockDim.x + threadIdx.x;
  if (idx >= (long)NN * HD) return;
  int n = (int)(idx / HD);
  int c = (int)(idx % HD);
  atomicAdd(&Pg[(long)batch[n]*HD + c], P[idx]);
}

__global__ void head_kernel(const float* __restrict__ Pg,
                            const float* __restrict__ hw, const float* __restrict__ hb,
                            float* __restrict__ out){
  int idx = blockIdx.x * blockDim.x + threadIdx.x;
  if (idx >= NGRAPH * NTGT) return;
  int g = idx / NTGT, t = idx % NTGT;
  float acc = hb[t];
  #pragma unroll 8
  for (int c = 0; c < HD; ++c) acc += Pg[g*HD + c] * hw[c*NTGT + t];
  out[idx] = acc;
}

// -------------------- fused WMMA linear: out = act(in@W [+b]) [+res] --------
//
// LDS layout (padded stride 40 halves = 80 B):
//   As[row][k]  row-contiguous  -> A fragment = 2x aligned ds_load_b128
//   Bs[col][k]  col-contiguous  -> B fragment = 2x aligned ds_load_b128
// Stride 40 keeps 16-B alignment for all chunks and spreads the 16 lanes of a
// half-wave across 16 distinct LDS banks (i*20 mod 64 is injective over 16).
// Out-of-range A rows are CLAMPED (not zero-filled): a padded row only
// affects its own (never-stored) output row, so no exec-mask branch needed.

template<int K, int N, bool DOSILU, bool HASBIAS, bool HASRES>
__global__ __launch_bounds__(256)
void linear_wmma_kernel(const float* __restrict__ in, const float* __restrict__ W,
                        const float* __restrict__ bias, const float* res,
                        float* out, int M){
  constexpr int KSTEP = 32;
  constexpr int NT = N / 16;
  constexpr int AST = 40;               // halves per row/col (32 + 8 pad)
  constexpr int AST8 = AST / 8;         // v8h per row/col
  __shared__ v8h As8[128 * AST8];       // 10 KB
  __shared__ v8h Bs8[N   * AST8];       // <= 20 KB
  _Float16* As = (_Float16*)As8;
  _Float16* Bs = (_Float16*)Bs8;

  const int tid  = threadIdx.x;
  const int wave = tid >> 5;
  const int lane = tid & 31;
  const int rowBase = blockIdx.x * 128;
  const int waveRow = rowBase + wave * 16;

  v8f acc[NT];
  #pragma unroll
  for (int i = 0; i < NT; ++i)
    #pragma unroll
    for (int j = 0; j < 8; ++j) acc[i][j] = 0.0f;

  for (int k0 = 0; k0 < K; k0 += KSTEP) {
    // ---- stage A panel (128 rows x 32 k), float4 loads, b64 LDS stores ----
    #pragma unroll
    for (int t = 0; t < 4; ++t) {
      int idx = tid + 256 * t;              // quad id, 1024 quads
      int rr  = idx >> 3;                   // 8 quads per row
      int k4  = (idx & 7) * 4;
      int gr  = rowBase + rr;
      if (gr >= M) gr = M - 1;              // clamp: padded rows never stored
      const float* ap = in + (long)gr * K + k0 + k4;
      float4 f = *(const float4*)ap;
      if (t == 0 && k0 + KSTEP < K)
        __builtin_prefetch(ap + KSTEP, 0, 3);   // global_prefetch next panel
      v4h hv;
      hv[0] = (_Float16)f.x; hv[1] = (_Float16)f.y;
      hv[2] = (_Float16)f.z; hv[3] = (_Float16)f.w;
      *(v4h*)(As + rr * AST + k4) = hv;
    }
    // ---- stage B panel transposed (N cols x 32 k) ----
    constexpr int BITER = N / 32;           // (8*N quads) / 256 threads
    #pragma unroll
    for (int t = 0; t < BITER; ++t) {
      int idx = tid + 256 * t;              // 0 .. 8N-1
      int nn  = idx % N;                    // lane-consecutive -> coalesced
      int k4  = (idx / N) * 4;
      const float* wp = W + (long)(k0 + k4) * N + nn;
      if (t == 0 && k0 + KSTEP < K)
        __builtin_prefetch(wp + (long)KSTEP * N, 0, 3);
      v4h hv;
      #pragma unroll
      for (int q = 0; q < 4; ++q)
        hv[q] = (_Float16)wp[(long)q * N];
      *(v4h*)(Bs + nn * AST + k4) = hv;
    }
    __syncthreads();

    // ---- A fragment: halves [kb..kb+7] and [kb+16..kb+23], kb = 0/8 ----
    const int half = lane >> 4;             // 0: lanes 0-15, 1: lanes 16-31
    const int m    = lane & 15;
    const v8h* arow = As8 + (wave * 16 + m) * AST8;
    v8h alo = arow[half];
    v8h ahi = arow[2 + half];
    v16h av = __builtin_shufflevector(alo, ahi,
        0,1,2,3,4,5,6,7,8,9,10,11,12,13,14,15);

    // ---- per 16-col tile: B fragment halves [kb..kb+7] and [kb+16..kb+23], kb = 0/8 ----
    #pragma unroll
    for (int nt = 0; nt < NT; ++nt) {
      const v8h* bcol = Bs8 + (nt * 16 + m) * AST8;
      v8h blo = bcol[half];
      v8h bhi = bcol[2 + half];
      v16h bv = __builtin_shufflevector(blo, bhi,
          0,1,2,3,4,5,6,7,8,9,10,11,12,13,14,15);
      acc[nt] = __builtin_amdgcn_wmma_f32_16x16x32_f16(
          false, av, false, bv, (short)0, acc[nt], false, false);
    }
    __syncthreads();
  }

  // ---- epilogue: C/D layout m = r + 8*(lane>>4), n = lane&15 ----
  const int mlo = (lane >> 4) * 8;
  const int nlo = lane & 15;
  #pragma unroll
  for (int nt = 0; nt < NT; ++nt) {
    int col = nt * 16 + nlo;
    #pragma unroll
    for (int r = 0; r < 8; ++r) {
      int row = waveRow + mlo + r;
      if (row < M) {
        float v = acc[nt][r];
        if (HASBIAS) v += bias[col];
        if (DOSILU)  v = silu_f(v);
        if (HASRES)  v += res[(long)row * N + col];
        out[(long)row * N + col] = v;
      }
    }
  }
}

// --------------------------- host orchestration ----------------------------

static inline unsigned gdiv(long n, int b){ long g = (n + b - 1) / b; return g < 1 ? 1u : (unsigned)g; }

template<int K, int N, bool DOSILU, bool HASBIAS, bool HASRES>
static void run_linear(hipStream_t s, const float* in, const float* W,
                       const float* b, const float* r, float* o, int M){
  linear_wmma_kernel<K, N, DOSILU, HASBIAS, HASRES>
      <<<dim3(gdiv(M, 128)), dim3(256), 0, s>>>(in, W, b, r, o, M);
}

static void zero_buf(hipStream_t s, float* p, long n){
  unsigned g = gdiv(n, 256); if (g > 4096u) g = 4096u;
  zero_kernel<<<dim3(g), dim3(256), 0, s>>>(p, n);
}

extern "C" void kernel_launch(void* const* d_in, const int* in_sizes, int n_in,
                              void* d_out, int out_size, void* d_ws, size_t ws_size,
                              hipStream_t stream){
  (void)n_in; (void)out_size; (void)ws_size;
  // pos, then params (sorted keys), then index arrays
  const float* pos        = (const float*)d_in[0];
  const float* emb_lin_b  = (const float*)d_in[1];
  const float* emb_lin_w  = (const float*)d_in[2];
  const float* emb_rbf_b  = (const float*)d_in[3];
  const float* emb_rbf_w  = (const float*)d_in[4];
  const float* emb_z      = (const float*)d_in[5];
  const float* freq       = (const float*)d_in[6];
  const float* head_b     = (const float*)d_in[7];
  const float* head_w     = (const float*)d_in[8];
  const float* int_as_b   = (const float*)d_in[9];
  const float* int_as_w   = (const float*)d_in[10];
  const float* int_bs_b   = (const float*)d_in[11];
  const float* int_bs_w   = (const float*)d_in[12];
  const float* int_down_w = (const float*)d_in[13];
  const float* int_ji_b   = (const float*)d_in[14];
  const float* int_ji_w   = (const float*)d_in[15];
  const float* int_kj_b   = (const float*)d_in[16];
  const float* int_kj_w   = (const float*)d_in[17];
  const float* int_lin_b  = (const float*)d_in[18];
  const float* int_lin_w  = (const float*)d_in[19];
  const float* int_rbf1_w = (const float*)d_in[20];
  const float* int_rbf2_w = (const float*)d_in[21];
  const float* int_sbf1_w = (const float*)d_in[22];
  const float* int_sbf2_w = (const float*)d_in[23];
  const float* int_up_w   = (const float*)d_in[24];
  const float* out_lins_b = (const float*)d_in[25];
  const float* out_lins_w = (const float*)d_in[26];
  const float* out_rbf_w  = (const float*)d_in[27];
  const float* out_up_w   = (const float*)d_in[28];
  const float* out_w      = (const float*)d_in[29];
  const int* zarr   = (const int*)d_in[30];
  const int* batch  = (const int*)d_in[31];
  const int* ei     = (const int*)d_in[32];
  const int* ej     = (const int*)d_in[33];
  const int* idx_i  = (const int*)d_in[34];
  const int* idx_j  = (const int*)d_in[35];
  const int* idx_k  = (const int*)d_in[36];
  const int* idx_kj = (const int*)d_in[37];
  const int* idx_ji = (const int*)d_in[38];

  const int E  = in_sizes[32];
  const int T  = in_sizes[34];
  const int NN = in_sizes[30];
  if (E <= 0 || NN <= 0) return;

  // workspace arena (floats, 64-float = 256-B aligned blocks for float4 loads)
  float* w = (float*)d_ws;
  size_t off = 0;
  auto alloc = [&](size_t n){ n = (n + 63) & ~(size_t)63;
                              float* p = w + off; off += n; return p; };
  float* rbf   = alloc((size_t)E * NRR);
  float* rbf_s = alloc((size_t)E * SBF_DIM);
  float* x     = alloc((size_t)E * HD);
  float* cat   = alloc((size_t)E * 384);    // reused below as xji|xkj|xdown|tacc
  float* xji   = cat;
  float* xkj   = cat + ((size_t)E * HD + 63 & ~(size_t)63);
  float* xdown = xkj + ((size_t)E * HD + 63 & ~(size_t)63);
  float* tacc  = xdown + ((size_t)E * IEMB + 63 & ~(size_t)63);
  float* h     = alloc((size_t)E * HD);
  float* t1    = alloc((size_t)E * HD);
  float* tn    = alloc((size_t)NN * HD);
  float* n256a = alloc((size_t)NN * OEMB);
  float* n256b = alloc((size_t)NN * OEMB);
  float* Pbuf  = alloc((size_t)NN * HD);
  float* Pg    = alloc((size_t)NGRAPH * HD);
  float* norm  = alloc(64);

  // basis precompute
  init_norm_kernel<<<dim3(1), dim3(64), 0, stream>>>(norm);
  edge_rbf_kernel<<<dim3(gdiv(E, 256)), dim3(256), 0, stream>>>(
      pos, ei, ej, freq, norm, rbf, rbf_s, E);

  // embedding: x = silu(cat @ emb_lin_w + b)
  build_cat_kernel<<<dim3(gdiv((long)E * 384, 256)), dim3(256), 0, stream>>>(
      rbf, zarr, ei, ej, emb_z, emb_rbf_w, emb_rbf_b, cat, E);
  run_linear<384, HD, true, true, false>(stream, cat, emb_lin_w, emb_lin_b, nullptr, x, E);

  zero_buf(stream, Pbuf, (long)NN * HD);

  // output block k applied to current x
  auto out_block = [&](int k){
    zero_buf(stream, tn, (long)NN * HD);
    out_scatter_kernel<<<dim3(gdiv((long)E * HD, 256)), dim3(256), 0, stream>>>(
        rbf, x, out_rbf_w + (size_t)k * NRR * HD, ei, tn, E);
    run_linear<HD, OEMB, false, false, false>(
        stream, tn, out_up_w + (size_t)k * HD * OEMB, nullptr, nullptr, n256a, NN);
    float* cur = n256a; float* oth = n256b;
    for (int l = 0; l < 3; ++l) {
      run_linear<OEMB, OEMB, true, true, false>(
          stream, cur, out_lins_w + ((size_t)k * 3 + l) * OEMB * OEMB,
          out_lins_b + ((size_t)k * 3 + l) * OEMB, nullptr, oth, NN);
      float* tmp = cur; cur = oth; oth = tmp;
    }
    run_linear<OEMB, HD, false, false, true>(
        stream, cur, out_w + (size_t)k * OEMB * HD, nullptr, Pbuf, Pbuf, NN);
  };

  out_block(0);

  for (int blk = 0; blk < NBLK; ++blk) {
    run_linear<HD, HD, true, true, false>(
        stream, x, int_ji_w + (size_t)blk * HD * HD, int_ji_b + (size_t)blk * HD, nullptr, xji, E);
    run_linear<HD, HD, true, true, false>(
        stream, x, int_kj_w + (size_t)blk * HD * HD, int_kj_b + (size_t)blk * HD, nullptr, xkj, E);
    gate_xkj_kernel<<<dim3(gdiv((long)E * HD, 256)), dim3(256), 0, stream>>>(
        rbf, int_rbf1_w + (size_t)blk * NRR * BAS, int_rbf2_w + (size_t)blk * BAS * HD, xkj, E);
    run_linear<HD, IEMB, true, false, false>(
        stream, xkj, int_down_w + (size_t)blk * HD * IEMB, nullptr, nullptr, xdown, E);

    zero_buf(stream, tacc, (long)E * IEMB);
    if (T > 0)
      triplet_kernel<<<dim3(gdiv(T, 256)), dim3(256), 0, stream>>>(
          pos, idx_i, idx_j, idx_k, idx_kj, idx_ji, rbf_s,
          int_sbf1_w + (size_t)blk * SBF_DIM * BAS,
          int_sbf2_w + (size_t)blk * BAS * IEMB, xdown, tacc, T);

    // h = silu(tacc @ up) + xji
    run_linear<IEMB, HD, true, false, true>(
        stream, tacc, int_up_w + (size_t)blk * IEMB * HD, nullptr, xji, h, E);

    // before-skip residual MLP (1 iter): h = h + silu(silu(h@W1+b1)@W2+b2)
    {
      const float* W1 = int_bs_w + (size_t)blk * 2 * HD * HD;
      const float* b1 = int_bs_b + (size_t)blk * 2 * HD;
      run_linear<HD, HD, true, true, false>(stream, h, W1, b1, nullptr, t1, E);
      run_linear<HD, HD, true, true, true >(stream, t1, W1 + HD * HD, b1 + HD, h, h, E);
    }
    // x = silu(h @ lin + b) + x
    run_linear<HD, HD, true, true, true>(
        stream, h, int_lin_w + (size_t)blk * HD * HD, int_lin_b + (size_t)blk * HD, x, x, E);
    // after-skip residual MLPs (2 iters)
    for (int rl = 0; rl < 2; ++rl) {
      const float* W1 = int_as_w + (((size_t)blk * 2 + rl) * 2) * HD * HD;
      const float* b1 = int_as_b + (((size_t)blk * 2 + rl) * 2) * HD;
      run_linear<HD, HD, true, true, false>(stream, x, W1, b1, nullptr, t1, E);
      run_linear<HD, HD, true, true, true >(stream, t1, W1 + HD * HD, b1 + HD, x, x, E);
    }
    out_block(blk + 1);
  }

  zero_buf(stream, Pg, (long)NGRAPH * HD);
  graph_pool_kernel<<<dim3(gdiv((long)NN * HD, 256)), dim3(256), 0, stream>>>(
      Pbuf, batch, Pg, NN);
  head_kernel<<<dim3(gdiv(NGRAPH * NTGT, 256)), dim3(256), 0, stream>>>(
      Pg, head_w, head_b, (float*)d_out);
}
